// WindowAttention_31155692765618
// MI455X (gfx1250) — compile-verified, matrix operand order checked by reading, not run on hardware
//
#include <hip/hip_runtime.h>
#include <stdint.h>

#define DIM      384
#define HEADS    12
#define NTOK     49
#define NPAD     64
#define HEAD_DIM 32
#define BWIN     4096
#define MTOT     (BWIN * NTOK)   // 200704, divisible by 128
#define NWMASK   64
#define QK_SCALE 0.17677669529663687f   // 32^-0.5

typedef __attribute__((ext_vector_type(16))) __bf16 v16bf;
typedef __attribute__((ext_vector_type(8)))  float  v8f;

union FragAB { v16bf v; unsigned u[8]; };
union FragC  { v8f   v; float    f[8]; };

// K-offset pattern for 16-bit A/B fragments (ISA 7.12.2, 16-bit A 16x32):
// VGPR r, lane-half h -> K = 16*(r/4) + 8*h + 2*(r%4)   (holds K, K+1 packed)
// For r=0..3 / r=4..7 this is a contiguous 8-element (16B) run -> b128 loads.
__device__ __forceinline__ int kpat(int r, int half) {
  return ((r >> 2) << 4) + (half << 3) + ((r & 3) << 1);
}

__device__ __forceinline__ unsigned short f2bf(float f) {
  unsigned u = __builtin_bit_cast(unsigned, f);
  u = (u + 0x7fffu + ((u >> 16) & 1u)) >> 16;   // round-to-nearest-even
  return (unsigned short)u;
}
__device__ __forceinline__ unsigned pack2bf(float a, float b) {
  return ((unsigned)f2bf(b) << 16) | (unsigned)f2bf(a);
}

// ---------------- prep: x -> bf16 (one-time, keeps GEMM K-loop conversion-free)
__global__ __launch_bounds__(256) void prep_xbf_kernel(
    const float* __restrict__ x, unsigned* __restrict__ xbf2) {
  size_t idx = (size_t)blockIdx.x * 256 + threadIdx.x;   // 4 floats per thread
  const size_t total = (size_t)MTOT * DIM / 4;
  if (idx < total) {
    float4 f = *(const float4*)(x + idx * 4);
    uint2 p;
    p.x = pack2bf(f.x, f.y);
    p.y = pack2bf(f.z, f.w);
    *(uint2*)(xbf2 + idx * 2) = p;
  }
}

// ---------------- prep: transpose+convert weights, gather bias table ---------
__global__ void prep_wT_kernel(const float* __restrict__ w,
                               unsigned short* __restrict__ wT, int K, int N) {
  int idx = blockIdx.x * 256 + threadIdx.x;        // idx = n*K + k
  if (idx < K * N) {
    int n = idx / K;
    int k = idx - n * K;
    wT[idx] = f2bf(w[k * N + n]);
  }
}

__global__ void prep_bias_kernel(const float* __restrict__ rb,
                                 const int* __restrict__ ridx,
                                 float* __restrict__ bt) {
  int idx = blockIdx.x * 256 + threadIdx.x;        // idx = h*2401 + i*49 + j
  if (idx < HEADS * NTOK * NTOK) {
    int h  = idx / (NTOK * NTOK);
    int ij = idx - h * NTOK * NTOK;
    bt[idx] = rb[ridx[ij] * HEADS + h];
  }
}

// ---------------- kernel 1: qkv = x @ qkv_w + b (bf16 WMMA, f32 acc) --------
__global__ __launch_bounds__(256) void qkv_gemm_kernel(
    const unsigned short* __restrict__ xbf, const unsigned short* __restrict__ wT,
    const float* __restrict__ bvec,
    unsigned short* __restrict__ qbuf, unsigned short* __restrict__ kbuf,
    unsigned short* __restrict__ vbuf) {
  const int lane = threadIdx.x & 31;
  const int wave = threadIdx.x >> 5;
  const int half = lane >> 4, ln = lane & 15;
  const int m0 = blockIdx.x * 128 + (wave >> 2) * 64;   // 1568 M-tiles
  const int n0 = blockIdx.y * 128 + (wave & 3) * 32;    // 9 N-tiles

  FragC c[4][2];
#pragma unroll
  for (int mi = 0; mi < 4; mi++)
#pragma unroll
    for (int ni = 0; ni < 2; ni++) c[mi][ni] = FragC{};

  for (int ks = 0; ks < DIM; ks += 32) {
    FragAB a[4], b[2];
#pragma unroll
    for (int mi = 0; mi < 4; mi++) {
      const unsigned short* ap = xbf + (size_t)(m0 + mi * 16 + ln) * DIM + ks;
      if (ks + 32 < DIM) __builtin_prefetch(ap + 32, 0, 1);
#pragma unroll
      for (int r = 0; r < 8; r++)
        a[mi].u[r] = *(const unsigned*)(ap + kpat(r, half));
    }
#pragma unroll
    for (int ni = 0; ni < 2; ni++) {
      const unsigned short* wp = wT + (size_t)(n0 + ni * 16 + ln) * DIM + ks;
#pragma unroll
      for (int r = 0; r < 8; r++)
        b[ni].u[r] = *(const unsigned*)(wp + kpat(r, half));
    }
#pragma unroll
    for (int mi = 0; mi < 4; mi++)
#pragma unroll
      for (int ni = 0; ni < 2; ni++)
        c[mi][ni].v = __builtin_amdgcn_wmma_f32_16x16x32_bf16(
            false, a[mi].v, false, b[ni].v, (short)0, c[mi][ni].v, false, false);
  }

  // epilogue: +bias, scale q, scatter into padded per-(b,h) bf16 buffers
#pragma unroll
  for (int mi = 0; mi < 4; mi++) {
    int rowb = m0 + mi * 16;
#pragma unroll
    for (int ni = 0; ni < 2; ni++) {
      int nc = n0 + ni * 16 + ln;               // 0..1151
      int part = nc / DIM;                      // 0=q 1=k 2=v
      int hh = (nc - part * DIM) >> 5;
      int d  = nc & 31;
      float bias = bvec[nc];
#pragma unroll
      for (int r = 0; r < 8; r++) {
        int row = rowb + r + 8 * half;
        int bw  = row / NTOK;
        int tok = row - bw * NTOK;
        float val = c[mi][ni].f[r] + bias;
        size_t bh = (size_t)bw * HEADS + hh;
        if (part == 0) {
          qbuf[(bh * NPAD + tok) * HEAD_DIM + d] = f2bf(val * QK_SCALE);
        } else if (part == 1) {
          kbuf[(bh * NPAD + tok) * HEAD_DIM + d] = f2bf(val);
        } else {
          vbuf[(bh * HEAD_DIM + d) * NPAD + tok] = f2bf(val);  // transposed
        }
      }
    }
  }
}

// ---------------- kernel 2: per-(window,head) attention ---------------------
__global__ __launch_bounds__(64) void attn_kernel(
    const unsigned short* __restrict__ qbuf, const unsigned short* __restrict__ kbuf,
    const unsigned short* __restrict__ vbuf, const float* __restrict__ bias_tab,
    const float* __restrict__ mask, unsigned short* __restrict__ abuf) {
  __shared__ float    S[2][NPAD * NPAD];
  __shared__ unsigned Pu[2][NPAD * NPAD / 2];     // bf16 P, dword-aligned
  const int lane = threadIdx.x & 31;
  const int wave = threadIdx.x >> 5;
  const int half = lane >> 4, ln = lane & 15;
  const int id = blockIdx.x * 2 + wave;           // 0..49151
  const int b = id / HEADS;
  const int h = id - b * HEADS;
  const unsigned short* qb = qbuf + (size_t)id * NPAD * HEAD_DIM;
  const unsigned short* kb = kbuf + (size_t)id * NPAD * HEAD_DIM;
  const unsigned short* vb = vbuf + (size_t)id * HEAD_DIM * NPAD;
  float* Sw = S[wave];
  unsigned short* Pw = (unsigned short*)Pu[wave];

  // S = q @ k^T   (64x64 padded, 16 WMMAs)
  FragAB aq[4], bk[4];
#pragma unroll
  for (int t = 0; t < 4; t++) {
    const unsigned short* qp = qb + (t * 16 + ln) * HEAD_DIM;
    const unsigned short* kp = kb + (t * 16 + ln) * HEAD_DIM;
#pragma unroll
    for (int r = 0; r < 8; r++) {
      int k = kpat(r, half);
      aq[t].u[r] = *(const unsigned*)(qp + k);
      bk[t].u[r] = *(const unsigned*)(kp + k);
    }
  }
#pragma unroll
  for (int mi = 0; mi < 4; mi++)
#pragma unroll
    for (int nj = 0; nj < 4; nj++) {
      FragC cfr = FragC{};
      cfr.v = __builtin_amdgcn_wmma_f32_16x16x32_bf16(
          false, aq[mi].v, false, bk[nj].v, (short)0, cfr.v, false, false);
#pragma unroll
      for (int r = 0; r < 8; r++)
        Sw[(mi * 16 + r + 8 * half) * NPAD + nj * 16 + ln] = cfr.f[r];
    }
  __syncthreads();

  // softmax rows (add relative bias + mask; padding masked to 0)
  const float* brow = bias_tab + (size_t)h * NTOK * NTOK;
  const float* mrow = mask + (size_t)(b & (NWMASK - 1)) * NTOK * NTOK;
  for (int pass = 0; pass < 2; pass++) {
    int i = lane + pass * 32;
    if (i < NTOK) {
      float mx = -3.0e38f;
      for (int j = 0; j < NTOK; j++) {
        float v = Sw[i * NPAD + j] + brow[i * NTOK + j] + mrow[i * NTOK + j];
        Sw[i * NPAD + j] = v;
        mx = fmaxf(mx, v);
      }
      float sum = 0.f;
      for (int j = 0; j < NTOK; j++) {
        float e = __expf(Sw[i * NPAD + j] - mx);
        Sw[i * NPAD + j] = e;
        sum += e;
      }
      float rinv = 1.0f / sum;
      for (int j = 0; j < NTOK; j++) Pw[i * NPAD + j] = f2bf(Sw[i * NPAD + j] * rinv);
      for (int j = NTOK; j < NPAD; j++) Pw[i * NPAD + j] = 0;
    } else if (i < NPAD) {
      for (int j = 0; j < NPAD; j++) Pw[i * NPAD + j] = 0;
    }
  }
  __syncthreads();

  // out = P @ V   (64x32, K=64 in two k=32 WMMA steps; 16 WMMAs)
#pragma unroll
  for (int mi = 0; mi < 4; mi++) {
    FragAB pa[2];
#pragma unroll
    for (int kc = 0; kc < 2; kc++)
#pragma unroll
      for (int r = 0; r < 8; r++)
        pa[kc].u[r] =
            *(const unsigned*)(Pw + (mi * 16 + ln) * NPAD + kc * 32 + kpat(r, half));
#pragma unroll
    for (int ni = 0; ni < 2; ni++) {
      FragC cfr = FragC{};
#pragma unroll
      for (int kc = 0; kc < 2; kc++) {
        FragAB bv;
#pragma unroll
        for (int r = 0; r < 8; r++)
          bv.u[r] =
              *(const unsigned*)(vb + (ni * 16 + ln) * NPAD + kc * 32 + kpat(r, half));
        cfr.v = __builtin_amdgcn_wmma_f32_16x16x32_bf16(
            false, pa[kc].v, false, bv.v, (short)0, cfr.v, false, false);
      }
#pragma unroll
      for (int r = 0; r < 8; r++) {
        int tok = mi * 16 + r + 8 * half;
        if (tok < NTOK)
          abuf[((size_t)b * NTOK + tok) * DIM + h * HEAD_DIM + ni * 16 + ln] =
              f2bf(cfr.f[r]);
      }
    }
  }
}

// ---------------- kernel 3: out = attn @ proj_w + b (fp32 out) --------------
__global__ __launch_bounds__(256) void proj_gemm_kernel(
    const unsigned short* __restrict__ A, const unsigned short* __restrict__ wT,
    const float* __restrict__ bvec, float* __restrict__ out) {
  const int lane = threadIdx.x & 31;
  const int wave = threadIdx.x >> 5;
  const int half = lane >> 4, ln = lane & 15;
  const int m0 = blockIdx.x * 128 + (wave >> 2) * 64;
  const int n0 = blockIdx.y * 128 + (wave & 3) * 32;    // 3 N-tiles

  FragC c[4][2];
#pragma unroll
  for (int mi = 0; mi < 4; mi++)
#pragma unroll
    for (int ni = 0; ni < 2; ni++) c[mi][ni] = FragC{};

  for (int ks = 0; ks < DIM; ks += 32) {
    FragAB a[4], b[2];
#pragma unroll
    for (int mi = 0; mi < 4; mi++) {
      const unsigned short* ap = A + (size_t)(m0 + mi * 16 + ln) * DIM + ks;
      if (ks + 32 < DIM) __builtin_prefetch(ap + 32, 0, 1);
#pragma unroll
      for (int r = 0; r < 8; r++)
        a[mi].u[r] = *(const unsigned*)(ap + kpat(r, half));
    }
#pragma unroll
    for (int ni = 0; ni < 2; ni++) {
      const unsigned short* wp = wT + (size_t)(n0 + ni * 16 + ln) * DIM + ks;
#pragma unroll
      for (int r = 0; r < 8; r++)
        b[ni].u[r] = *(const unsigned*)(wp + kpat(r, half));
    }
#pragma unroll
    for (int mi = 0; mi < 4; mi++)
#pragma unroll
      for (int ni = 0; ni < 2; ni++)
        c[mi][ni].v = __builtin_amdgcn_wmma_f32_16x16x32_bf16(
            false, a[mi].v, false, b[ni].v, (short)0, c[mi][ni].v, false, false);
  }
#pragma unroll
  for (int mi = 0; mi < 4; mi++) {
#pragma unroll
    for (int ni = 0; ni < 2; ni++) {
      int nc = n0 + ni * 16 + ln;
      float bias = bvec[nc];
#pragma unroll
      for (int r = 0; r < 8; r++) {
        int row = m0 + mi * 16 + r + 8 * half;
        out[(size_t)row * DIM + nc] = c[mi][ni].f[r] + bias;
      }
    }
  }
}

// ---------------- host launcher ---------------------------------------------
extern "C" void kernel_launch(void* const* d_in, const int* in_sizes, int n_in,
                              void* d_out, int out_size, void* d_ws, size_t ws_size,
                              hipStream_t stream) {
  (void)in_sizes; (void)n_in; (void)out_size; (void)ws_size;
  const float* x    = (const float*)d_in[0];
  const float* mask = (const float*)d_in[1];
  const int*   ridx = (const int*)d_in[2];
  const float* qkvw = (const float*)d_in[3];
  const float* qkvb = (const float*)d_in[4];
  const float* pw   = (const float*)d_in[5];
  const float* pb   = (const float*)d_in[6];
  const float* rb   = (const float*)d_in[7];
  float* out = (float*)d_out;

  char* ws = (char*)d_ws;
  size_t cur = 0;
  auto wsalloc = [&](size_t bytes) -> char* {
    char* p = ws + cur;
    cur = (cur + bytes + 255) & ~(size_t)255;
    return p;
  };
  const size_t qkvTbytes  = (size_t)3 * DIM * DIM * 2;           //  884,736
  const size_t projTbytes = (size_t)DIM * DIM * 2;               //  294,912
  const size_t biasBytes  = (size_t)HEADS * NTOK * NTOK * 4;     //  115,248
  const size_t xbfBytes   = (size_t)MTOT * DIM * 2;              // ~147 MiB
  const size_t headBytes  = (size_t)BWIN * HEADS * NPAD * HEAD_DIM * 2;  // 192 MiB
  const size_t abufBytes  = (size_t)MTOT * DIM * 2;              // ~147 MiB

  unsigned short* qkv_wT   = (unsigned short*)wsalloc(qkvTbytes);
  unsigned short* proj_wT  = (unsigned short*)wsalloc(projTbytes);
  float*          bias_tab = (float*)wsalloc(biasBytes);
  unsigned short* xbf  = (unsigned short*)wsalloc(xbfBytes);
  unsigned short* qbuf = (unsigned short*)wsalloc(headBytes);
  unsigned short* kbuf = (unsigned short*)wsalloc(headBytes);
  unsigned short* vbuf = (unsigned short*)wsalloc(headBytes);
  unsigned short* abuf = (unsigned short*)wsalloc(abufBytes);

  // zero v so the 49->64 token padding contributes exact zeros through P@V
  hipMemsetAsync(vbuf, 0, headBytes, stream);

  prep_xbf_kernel<<<(unsigned)(((size_t)MTOT * DIM / 4 + 255) / 256), 256, 0,
                    stream>>>(x, (unsigned*)xbf);
  prep_wT_kernel<<<(3 * DIM * DIM + 255) / 256, 256, 0, stream>>>(
      qkvw, qkv_wT, DIM, 3 * DIM);
  prep_wT_kernel<<<(DIM * DIM + 255) / 256, 256, 0, stream>>>(
      pw, proj_wT, DIM, DIM);
  prep_bias_kernel<<<(HEADS * NTOK * NTOK + 255) / 256, 256, 0, stream>>>(
      rb, ridx, bias_tab);

  qkv_gemm_kernel<<<dim3(MTOT / 128, (3 * DIM) / 128), 256, 0, stream>>>(
      xbf, qkv_wT, qkvb, qbuf, kbuf, vbuf);

  attn_kernel<<<(BWIN * HEADS) / 2, 64, 0, stream>>>(
      qbuf, kbuf, vbuf, bias_tab, mask, abuf);

  proj_gemm_kernel<<<dim3(MTOT / 128, DIM / 128), 256, 0, stream>>>(
      abuf, proj_wT, pb, out);
}
